// BusStopPredictor_5050881540303
// MI455X (gfx1250) — matrix operand order repaired and checked by reading, not compiled, not run on hardware
//
#include <hip/hip_runtime.h>

#define N_NODES 50000
#define N_EDGES 800000
#define IN_C 32
#define HID 64
#define BN_EPS 1e-5f

typedef __attribute__((ext_vector_type(16))) _Float16 v16h;
typedef __attribute__((ext_vector_type(8)))  float    v8f;

// ---------------------------------------------------------------------------
// Utility: zero a float region
// ---------------------------------------------------------------------------
__global__ void zero_f32_kernel(float* __restrict__ p, int count) {
    int i = blockIdx.x * blockDim.x + threadIdx.x;
    if (i < count) p[i] = 0.0f;
}

// ---------------------------------------------------------------------------
// Degree: one thread per edge, f32 atomic add of 1.0 to deg[dst]
// ---------------------------------------------------------------------------
__global__ void degree_kernel(const long long* __restrict__ dst,
                              float* __restrict__ deg) {
    int e = blockIdx.x * blockDim.x + threadIdx.x;   // grid sized exactly E
    atomicAdd(&deg[(int)dst[e]], 1.0f);
}

__global__ void inv_deg_kernel(float* __restrict__ deg) {
    int i = blockIdx.x * blockDim.x + threadIdx.x;
    if (i < N_NODES) deg[i] = 1.0f / fmaxf(deg[i], 1.0f);
}

// ---------------------------------------------------------------------------
// Scatter-sum of features over edges.
// Thread t -> edge e = t/C, channel c = t%C.  With wave32, one wave covers
// 32 consecutive channels of one edge: coalesced gather from feat[src] and a
// contiguous burst of f32 atomics into agg[dst] (both L2-resident).
// ---------------------------------------------------------------------------
template <int C>
__global__ void scatter_sum_kernel(const long long* __restrict__ src,
                                   const long long* __restrict__ dst,
                                   const float* __restrict__ feat,
                                   float* __restrict__ agg) {
    int t = blockIdx.x * blockDim.x + threadIdx.x;   // grid sized exactly E*C
    int e = t / C;
    int c = t % C;
    int s = (int)src[e];
    int d = (int)dst[e];
    atomicAdd(&agg[d * C + c], feat[s * C + c]);
}

// ---------------------------------------------------------------------------
// Fused SAGE layer:  out = relu( bn( (agg*deg_inv) @ Wl + bias + root @ Wr ) )
//
// Block = 128 threads = 4 waves. Block b owns output rows [16b, 16b+16);
// wave w owns output cols [16w, 16w+16). Each wave issues 2*KS
// v_wmma_f32_16x16x32_f16 ops (KS = CIN/32) with f32 accumulation.
//
// Stage 1: the block's two A tiles (agg rows, root rows: each 16 x CIN,
//          contiguous in memory) are copied into LDS with coalesced b64
//          global loads. LDS row stride is CIN+2 floats so the 16 rows of a
//          half-wave map to 16 distinct banks -> conflict-free ds_load_b64
//          in the WMMA-layout read phase.
// Stage 2: lanes read their A fragments from LDS per the ISA 16-bit 16x32
//          A layout (lane holds row M = lane&15; VGPR i holds K-pair
//          i<4 ? half*8+2i : 16+half*8+2(i-4)); B cols are gathered from the
//          (L0-hot) row-major weights; WMMA; fused bias+BN+ReLU epilogue.
//
// In-place safe (out may alias aggIn): every global read of the aliased A
// tile happens before the single __syncthreads(); stores happen after, and a
// block only writes the rows it owns.
// ---------------------------------------------------------------------------
template <int CIN>
__global__ void __launch_bounds__(128)
sage_layer_kernel(const float* __restrict__ aggIn,   // [N, CIN]
                  const float* __restrict__ rootIn,  // [N, CIN]
                  const float* __restrict__ degInv,  // [N]
                  const float* __restrict__ Wl,      // [CIN, 64] row-major
                  const float* __restrict__ biasL,   // [64]
                  const float* __restrict__ Wr,      // [CIN, 64] row-major
                  const float* __restrict__ bnG, const float* __restrict__ bnB,
                  const float* __restrict__ bnM, const float* __restrict__ bnV,
                  float* __restrict__ outp)          // [N, 64]
{
    constexpr int KS = CIN / 32;
    constexpr int S  = CIN + 2;              // padded LDS row stride (floats)
    __shared__ float lds_agg [16 * S];
    __shared__ float lds_root[16 * S];

    const int tid = threadIdx.x;

    // ---- Stage 1: coalesced copy of the two contiguous 16xCIN A tiles ----
    {
        const float* aggBase  = aggIn  + blockIdx.x * (16 * CIN);
        const float* rootBase = rootIn + blockIdx.x * (16 * CIN);
        constexpr int UNITS = 16 * CIN / 2;  // float2 units per matrix
        constexpr int ITER  = UNITS / 128;   // 2 (CIN=32) or 4 (CIN=64)
#pragma unroll
        for (int k = 0; k < ITER; ++k) {
            const int u   = tid + k * 128;
            const int row = u / (CIN / 2);
            const int col = (u % (CIN / 2)) * 2;
            const float2 pa = *(const float2*)(aggBase  + row * CIN + col);
            const float2 pr = *(const float2*)(rootBase + row * CIN + col);
            *(float2*)(&lds_agg [row * S + col]) = pa;
            *(float2*)(&lds_root[row * S + col]) = pr;
        }
    }
    __syncthreads();   // staging done; aliased global reads all complete

    // ---- Stage 2: WMMA fragments ----
    const int lane = tid & 31;
    const int wave = tid >> 5;
    const int n    = lane & 15;
    const int half = lane >> 4;
    const int rowA = blockIdx.x * 16 + n;    // A row this lane owns
    const int colB = (wave << 4) + n;        // B col / output col

    const float scale = degInv[rowA];

    v16h a1[KS], a2[KS], wl[KS], wr[KS];
#pragma unroll
    for (int ks = 0; ks < KS; ++ks) {
        const int ko = ks * 32;
#pragma unroll
        for (int i = 0; i < 8; ++i) {
            // A layout K offset for VGPR i
            const int kk = ko + ((i & 4) << 2) + (half << 3) + ((i & 3) << 1);
            const float2 p = *(const float2*)(&lds_agg [n * S + kk]);
            const float2 q = *(const float2*)(&lds_root[n * S + kk]);
            a1[ks][2 * i]     = (_Float16)(p.x * scale);
            a1[ks][2 * i + 1] = (_Float16)(p.y * scale);
            a2[ks][2 * i]     = (_Float16)q.x;
            a2[ks][2 * i + 1] = (_Float16)q.y;
            // B layout K index for VGPR i (same K order as A)
            const int k = ko + ((i & 4) << 2) + (half << 3) + ((i & 3) << 1);
            wl[ks][2 * i]     = (_Float16)Wl[k * HID + colB];
            wl[ks][2 * i + 1] = (_Float16)Wl[(k + 1) * HID + colB];
            wr[ks][2 * i]     = (_Float16)Wr[k * HID + colB];
            wr[ks][2 * i + 1] = (_Float16)Wr[(k + 1) * HID + colB];
        }
    }

    v8f c = {};
#pragma unroll
    for (int ks = 0; ks < KS; ++ks) {
        c = __builtin_amdgcn_wmma_f32_16x16x32_f16(
                false, a1[ks], false, wl[ks], (short)0, c, false, false);
        c = __builtin_amdgcn_wmma_f32_16x16x32_f16(
                false, a2[ks], false, wr[ks], (short)0, c, false, false);
    }

    // Fused bias + batchnorm(eval) + relu epilogue
    const float bScale = bnG[colB] * rsqrtf(bnV[colB] + BN_EPS);
    const float bMean  = bnM[colB];
    const float bBeta  = bnB[colB];
    const float bs     = biasL[colB];
    float* outBase = outp + (blockIdx.x * 16) * HID + colB;
#pragma unroll
    for (int j = 0; j < 8; ++j) {
        const int mrow = (half << 3) + j;     // D row for element j
        float val = c[j] + bs;
        val = (val - bMean) * bScale + bBeta;
        outBase[mrow * HID] = fmaxf(val, 0.0f);
    }
}

// ---------------------------------------------------------------------------
// Final linear 64 -> 1
// ---------------------------------------------------------------------------
__global__ void final_lin_kernel(const float* __restrict__ h,
                                 const float* __restrict__ Wlin,
                                 const float* __restrict__ blin,
                                 float* __restrict__ out) {
    int i = blockIdx.x * blockDim.x + threadIdx.x;
    if (i >= N_NODES) return;
    const float4* hp = (const float4*)(h + i * HID);
    const float4* wp = (const float4*)Wlin;
    float acc = 0.0f;
#pragma unroll
    for (int j = 0; j < 16; ++j) {
        float4 a = hp[j], w = wp[j];
        acc += a.x * w.x + a.y * w.y + a.z * w.z + a.w * w.w;
    }
    out[i] = acc + blin[0];
}

// ---------------------------------------------------------------------------
// Launch
// ---------------------------------------------------------------------------
static inline int cdiv_i(int a, int b) { return (a + b - 1) / b; }

extern "C" void kernel_launch(void* const* d_in, const int* in_sizes, int n_in,
                              void* d_out, int out_size, void* d_ws, size_t ws_size,
                              hipStream_t stream) {
    const float*     x     = (const float*)d_in[0];
    const long long* ei    = (const long long*)d_in[1];   // int64 [2, E]
    const float*     W1l   = (const float*)d_in[2];
    const float*     b1l   = (const float*)d_in[3];
    const float*     W1r   = (const float*)d_in[4];
    const float*     bn1_g = (const float*)d_in[5];
    const float*     bn1_b = (const float*)d_in[6];
    const float*     bn1_m = (const float*)d_in[7];
    const float*     bn1_v = (const float*)d_in[8];
    const float*     W2l   = (const float*)d_in[9];
    const float*     b2l   = (const float*)d_in[10];
    const float*     W2r   = (const float*)d_in[11];
    const float*     bn2_g = (const float*)d_in[12];
    const float*     bn2_b = (const float*)d_in[13];
    const float*     bn2_m = (const float*)d_in[14];
    const float*     bn2_v = (const float*)d_in[15];
    const float*     Wlin  = (const float*)d_in[16];
    const float*     blin  = (const float*)d_in[17];
    float*           out   = (float*)d_out;

    const long long* esrc = ei;
    const long long* edst = ei + N_EDGES;

    // Workspace layout (floats): deg[N] | agg[N*64] | h[N*64]   (~25.8 MB)
    float* ws  = (float*)d_ws;
    float* deg = ws;
    float* agg = ws + N_NODES;
    float* h   = ws + N_NODES + N_NODES * HID;

    // 1. zero deg + agg (contiguous)
    {
        int cnt = N_NODES * (1 + HID);
        zero_f32_kernel<<<cdiv_i(cnt, 256), 256, 0, stream>>>(deg, cnt);
    }
    // 2. degree (E divides 256 exactly)
    degree_kernel<<<N_EDGES / 256, 256, 0, stream>>>(edst, deg);
    // 3. deg -> 1/max(deg,1)
    inv_deg_kernel<<<cdiv_i(N_NODES, 256), 256, 0, stream>>>(deg);
    // 4. scatter x[src] into agg (stride 32); E*32 divides 256 exactly
    scatter_sum_kernel<IN_C><<<(N_EDGES * IN_C) / 256, 256, 0, stream>>>(
        esrc, edst, x, agg);
    // 5. layer 1 fused GEMM (WMMA) + BN + ReLU -> h
    sage_layer_kernel<IN_C><<<N_NODES / 16, 128, 0, stream>>>(
        agg, x, deg, W1l, b1l, W1r, bn1_g, bn1_b, bn1_m, bn1_v, h);
    // 6. re-zero agg for layer 2
    {
        int cnt = N_NODES * HID;
        zero_f32_kernel<<<cdiv_i(cnt, 256), 256, 0, stream>>>(agg, cnt);
    }
    // 7. scatter h[src] into agg (stride 64); E*64 divides 256 exactly
    scatter_sum_kernel<HID><<<(N_EDGES * HID) / 256, 256, 0, stream>>>(
        esrc, edst, h, agg);
    // 8. layer 2 fused GEMM (WMMA, K=64 -> 2 k-steps) + BN + ReLU, in-place
    sage_layer_kernel<HID><<<N_NODES / 16, 128, 0, stream>>>(
        agg, h, deg, W2l, b2l, W2r, bn2_g, bn2_b, bn2_m, bn2_v, agg);
    // 9. final 64->1 linear
    final_lin_kernel<<<cdiv_i(N_NODES, 256), 256, 0, stream>>>(agg, Wlin, blin, out);
}
